// BiGRU_73985106641018
// MI455X (gfx1250) — matrix operand order repaired and checked by reading, not compiled, not run on hardware
//
#include <hip/hip_runtime.h>
#include <hip/hip_bf16.h>
#include <math.h>

#define SEQ   512
#define BATCH 64
#define INDIM 512
#define HID   512
#define G3    1536   // 3*HID

typedef __bf16 bf16_t;
typedef __attribute__((ext_vector_type(16))) __bf16 v16bf;
typedef __attribute__((ext_vector_type(8)))  __bf16 v8bf;
typedef __attribute__((ext_vector_type(8)))  float  v8f;

// ---------------------------------------------------------------------------
// Split an f32 array into bf16 hi + bf16 lo (residual) arrays (weights, x).
// ---------------------------------------------------------------------------
__global__ void split_bf16_kernel(const float* __restrict__ w,
                                  bf16_t* __restrict__ hi,
                                  bf16_t* __restrict__ lo, int n) {
  int i = blockIdx.x * blockDim.x + threadIdx.x;
  if (i < n) {
    float f = w[i];
    bf16_t h = (bf16_t)f;
    hi[i] = h;
    lo[i] = (bf16_t)(f - (float)h);
  }
}

// ---------------------------------------------------------------------------
// A-fragment loader from pre-split bf16 activations (row-major, row=M, K).
// Contiguous k order (this code's choice, matched by B): lane l holds row (l&15); lanes<16 own K=kc+0..15,
// lanes>=16 own K=kc+16..31 (matches B). Two 16B loads, no VALU.
// ---------------------------------------------------------------------------
union V16U { v16bf v; v8bf h[2]; };

__device__ __forceinline__ v16bf load_act_frag(const bf16_t* __restrict__ base,
                                               int ldk, int row, int kc,
                                               int lane) {
  const int koff0 = (lane >> 4) << 4;  // 0 or 16
  const bf16_t* p = base + (size_t)row * ldk + kc + koff0;
  V16U u;
  u.h[0] = *(const v8bf*)(p);
  u.h[1] = *(const v8bf*)(p + 8);
  return u.v;
}

// ---------------------------------------------------------------------------
// B-fragment: operand is W^T (KxN); lane n<16 holds col n (= row rowBase+n of
// W) K=kc+0..15, lane n+16 holds K=kc+16..31. One 32B load per lane.
// ---------------------------------------------------------------------------
__device__ __forceinline__ v16bf load_b_frag(const bf16_t* __restrict__ w,
                                             int ldk, int rowBase, int kc,
                                             int lane) {
  const int n = lane & 15;
  const int koff = (lane >> 4) << 4;  // 0 or 16
  return *(const v16bf*)(w + (size_t)(rowBase + n) * ldk + kc + koff);
}

__device__ __forceinline__ v8f wmma_bf16(v16bf a, v16bf b, v8f c) {
  return __builtin_amdgcn_wmma_f32_16x16x32_bf16(false, a, false, b,
                                                 (short)0, c, false, false);
}

// ---------------------------------------------------------------------------
// Big parallel GEMM: G[r, j3] = act[srow(r)] . W[j3] + bias[j3]
// act: (SEQ*BATCH, K) bf16 hi/lo; W: (1536, K) bf16 hi/lo; G: f32.
// flip=1 applies the reference's batch-axis flip inside each t-slice.
// grid = (2048, 24), block = 128 (4 waves; wave = one 16x16 tile).
// ---------------------------------------------------------------------------
__global__ __launch_bounds__(128) void gi_gemm_kernel(
    const bf16_t* __restrict__ a_hi, const bf16_t* __restrict__ a_lo, int K,
    int flip, const bf16_t* __restrict__ w_hi, const bf16_t* __restrict__ w_lo,
    const float* __restrict__ bias, float* __restrict__ G) {
  const int lane = threadIdx.x & 31;
  const int wv = threadIdx.x >> 5;
  const int r0 = blockIdx.x * 16;
  const int j0 = (blockIdx.y * 4 + wv) * 16;

  const int m = lane & 15;
  const int r = r0 + m;
  const int t = r >> 6;
  const int b = r & 63;
  const int srow = flip ? ((t << 6) + (63 - b)) : r;

  v8f acc = {};
  for (int kc = 0; kc < K; kc += 32) {
    v16bf ahi = load_act_frag(a_hi, K, srow, kc, lane);
    v16bf alo = load_act_frag(a_lo, K, srow, kc, lane);
    v16bf bhi = load_b_frag(w_hi, K, j0, kc, lane);
    v16bf blo = load_b_frag(w_lo, K, j0, kc, lane);
    acc = wmma_bf16(ahi, bhi, acc);
    acc = wmma_bf16(ahi, blo, acc);
    acc = wmma_bf16(alo, bhi, acc);
  }
  const int n = lane & 15;
  const float bv = bias[j0 + n];
#pragma unroll
  for (int v = 0; v < 8; ++v) {
    const int rr = r0 + v + ((lane >> 4) << 3);
    G[(size_t)rr * G3 + j0 + n] = acc[v] + bv;
  }
}

// ---------------------------------------------------------------------------
// One GRU time-step, both directions (blockIdx.y = dir). PRE=true: gi was
// precomputed into G (bias folded in); only the gh GEMM runs sequentially.
// PRE=false: fused gi+gh GEMMs from pre-split activations.
// grid = (32, 2), block = 128; one wave = one 16(batch) x 16(hidden) tile.
// ---------------------------------------------------------------------------
template <bool PRE, int IN_DIM>
__global__ __launch_bounds__(128) void gru_step_kernel(
    const bf16_t* __restrict__ x_hi, const bf16_t* __restrict__ x_lo,
    const float* __restrict__ giF, const float* __restrict__ giB,
    const bf16_t* __restrict__ wihHiF, const bf16_t* __restrict__ wihLoF,
    const bf16_t* __restrict__ whhHiF, const bf16_t* __restrict__ whhLoF,
    const float* __restrict__ bihF, const float* __restrict__ bhhF,
    const bf16_t* __restrict__ wihHiB, const bf16_t* __restrict__ wihLoB,
    const bf16_t* __restrict__ whhHiB, const bf16_t* __restrict__ whhLoB,
    const float* __restrict__ bihB, const float* __restrict__ bhhB,
    const float* __restrict__ hin32F, const bf16_t* __restrict__ hinHF,
    const bf16_t* __restrict__ hinLF, float* __restrict__ hout32F,
    bf16_t* __restrict__ houtHF, bf16_t* __restrict__ houtLF,
    const float* __restrict__ hin32B, const bf16_t* __restrict__ hinHB,
    const bf16_t* __restrict__ hinLB, float* __restrict__ hout32B,
    bf16_t* __restrict__ houtHB, bf16_t* __restrict__ houtLB,
    float* __restrict__ out32, bf16_t* __restrict__ outH,
    bf16_t* __restrict__ outL, int outStride) {
  const int lane = threadIdx.x & 31;
  const int wv = threadIdx.x >> 5;
  const int gw = blockIdx.x * 4 + wv;  // 0..127
  const int b0 = (gw & 3) * 16;
  const int j0 = (gw >> 2) * 16;
  const int dir = blockIdx.y;

  const bf16_t* whhHi = dir ? whhHiB : whhHiF;
  const bf16_t* whhLo = dir ? whhLoB : whhLoF;
  const float* bhh = dir ? bhhB : bhhF;
  const float* hin32 = dir ? hin32B : hin32F;
  const bf16_t* hinH = dir ? hinHB : hinHF;
  const bf16_t* hinL = dir ? hinLB : hinLF;
  float* hout32 = dir ? hout32B : hout32F;
  bf16_t* houtH = dir ? houtHB : houtHF;
  bf16_t* houtL = dir ? houtLB : houtLF;
  const int outOff = dir ? HID : 0;

  const int mrow = lane & 15;
  const int hrow = b0 + mrow;

  const v8f zacc = {};
  v8f acc_h[3] = {zacc, zacc, zacc};
  v8f acc_i[3] = {zacc, zacc, zacc};

  if constexpr (!PRE) {
    // gi = x @ Wih^T (K = IN_DIM); backward cell reads x with batch flipped
    const bf16_t* wihHi = dir ? wihHiB : wihHiF;
    const bf16_t* wihLo = dir ? wihLoB : wihLoF;
    const int xrow = dir ? (BATCH - 1) - (b0 + mrow) : (b0 + mrow);
    for (int kc = 0; kc < IN_DIM; kc += 32) {
      v16bf ahi = load_act_frag(x_hi, IN_DIM, xrow, kc, lane);
      v16bf alo = load_act_frag(x_lo, IN_DIM, xrow, kc, lane);
#pragma unroll
      for (int g = 0; g < 3; ++g) {
        v16bf bhi = load_b_frag(wihHi, IN_DIM, g * HID + j0, kc, lane);
        v16bf blo = load_b_frag(wihLo, IN_DIM, g * HID + j0, kc, lane);
        acc_i[g] = wmma_bf16(ahi, bhi, acc_i[g]);
        acc_i[g] = wmma_bf16(ahi, blo, acc_i[g]);
        acc_i[g] = wmma_bf16(alo, bhi, acc_i[g]);
      }
    }
  }

  // gh = h @ Whh^T (K = HID)
  for (int kc = 0; kc < HID; kc += 32) {
    v16bf ahi = load_act_frag(hinH, HID, hrow, kc, lane);
    v16bf alo = load_act_frag(hinL, HID, hrow, kc, lane);
#pragma unroll
    for (int g = 0; g < 3; ++g) {
      v16bf bhi = load_b_frag(whhHi, HID, g * HID + j0, kc, lane);
      v16bf blo = load_b_frag(whhLo, HID, g * HID + j0, kc, lane);
      acc_h[g] = wmma_bf16(ahi, bhi, acc_h[g]);
      acc_h[g] = wmma_bf16(ahi, blo, acc_h[g]);
      acc_h[g] = wmma_bf16(alo, bhi, acc_h[g]);
    }
  }

  // Epilogue. C/D layout: VGPR v, lane l -> M = v + 8*(l>=16), N = l&15.
  const int n = lane & 15;
  const int jj = j0 + n;
  const float* gi = dir ? giB : giF;
  const float* bih = dir ? bihB : bihF;
  float bir = 0.f, biz = 0.f, bin = 0.f;
  if constexpr (!PRE) {
    bir = bih[jj]; biz = bih[jj + HID]; bin = bih[jj + 2 * HID];
  }
  const float bhr = bhh[jj], bhz = bhh[jj + HID], bhn = bhh[jj + 2 * HID];
#pragma unroll
  for (int v = 0; v < 8; ++v) {
    const int m = v + ((lane >> 4) << 3);
    const int b = b0 + m;
    float ir, iz, in_;
    if constexpr (PRE) {
      const float* gr = gi + (size_t)b * G3;  // bias already folded in
      ir = gr[jj]; iz = gr[jj + HID]; in_ = gr[jj + 2 * HID];
    } else {
      ir = acc_i[0][v] + bir;
      iz = acc_i[1][v] + biz;
      in_ = acc_i[2][v] + bin;
    }
    const float hr = acc_h[0][v] + bhr;
    const float hz = acc_h[1][v] + bhz;
    const float hn = acc_h[2][v] + bhn;
    const float r = 1.0f / (1.0f + __expf(-(ir + hr)));
    const float z = 1.0f / (1.0f + __expf(-(iz + hz)));
    const float ng = tanhf(in_ + r * hn);
    const float hp = hin32[(size_t)b * HID + jj];
    const float hv = (1.0f - z) * ng + z * hp;
    hout32[(size_t)b * HID + jj] = hv;
    const bf16_t hh = (bf16_t)hv;
    houtH[(size_t)b * HID + jj] = hh;
    houtL[(size_t)b * HID + jj] = (bf16_t)(hv - (float)hh);
    const size_t oidx = (size_t)b * outStride + outOff + jj;
    if (out32) out32[oidx] = hv;
    if (outH) {
      outH[oidx] = hh;
      outL[oidx] = (bf16_t)(hv - (float)hh);
    }
  }
}

// ---------------------------------------------------------------------------
// Host orchestration.
// ---------------------------------------------------------------------------
extern "C" void kernel_launch(void* const* d_in, const int* in_sizes, int n_in,
                              void* d_out, int out_size, void* d_ws,
                              size_t ws_size, hipStream_t stream) {
  (void)in_sizes; (void)n_in; (void)out_size;
  const float* x = (const float*)d_in[0];
  const float* W[8] = {(const float*)d_in[1],  (const float*)d_in[2],
                       (const float*)d_in[5],  (const float*)d_in[6],
                       (const float*)d_in[9],  (const float*)d_in[10],
                       (const float*)d_in[13], (const float*)d_in[14]};
  const int wn[8] = {G3 * INDIM, G3 * HID, G3 * INDIM, G3 * HID,
                     G3 * 2 * HID, G3 * HID, G3 * 2 * HID, G3 * HID};
  const float* bih_f0 = (const float*)d_in[3];
  const float* bhh_f0 = (const float*)d_in[4];
  const float* bih_b0 = (const float*)d_in[7];
  const float* bhh_b0 = (const float*)d_in[8];
  const float* bih_f1 = (const float*)d_in[11];
  const float* bhh_f1 = (const float*)d_in[12];
  const float* bih_b1 = (const float*)d_in[15];
  const float* bhh_b1 = (const float*)d_in[16];

  char* ws = (char*)d_ws;
  size_t off = 0;
  auto carve = [&](size_t bytes) -> char* {
    off = (off + 255) & ~(size_t)255;
    char* p = ws + off;
    off += bytes;
    return p;
  };

  bf16_t *whi[8], *wlo[8];
  for (int i = 0; i < 8; ++i) {
    whi[i] = (bf16_t*)carve((size_t)wn[i] * sizeof(bf16_t));
    wlo[i] = (bf16_t*)carve((size_t)wn[i] * sizeof(bf16_t));
  }
  const size_t XN = (size_t)SEQ * BATCH * INDIM;
  bf16_t* xH = (bf16_t*)carve(XN * sizeof(bf16_t));
  bf16_t* xL = (bf16_t*)carve(XN * sizeof(bf16_t));
  const size_t ON = (size_t)SEQ * BATCH * 2 * HID;
  bf16_t* o0H = (bf16_t*)carve(ON * sizeof(bf16_t));
  bf16_t* o0L = (bf16_t*)carve(ON * sizeof(bf16_t));
  const size_t BH = (size_t)BATCH * HID;
  float* h32 = (float*)carve(8 * BH * sizeof(float));
  bf16_t* hH = (bf16_t*)carve(8 * BH * sizeof(bf16_t));
  bf16_t* hL = (bf16_t*)carve(8 * BH * sizeof(bf16_t));
  // G buffers only used if workspace is large enough
  const size_t GN = (size_t)SEQ * BATCH * G3;
  float* Gf = (float*)carve(GN * sizeof(float));
  float* Gb = (float*)carve(GN * sizeof(float));
  const bool big = (ws_size >= off);

  auto H32 = [&](int s, int p) { return h32 + (size_t)(2 * s + p) * BH; };
  auto HHp = [&](int s, int p) { return hH + (size_t)(2 * s + p) * BH; };
  auto HLp = [&](int s, int p) { return hL + (size_t)(2 * s + p) * BH; };

  for (int i = 0; i < 8; ++i) {
    split_bf16_kernel<<<(wn[i] + 255) / 256, 256, 0, stream>>>(W[i], whi[i],
                                                               wlo[i], wn[i]);
  }
  split_bf16_kernel<<<(int)((XN + 255) / 256), 256, 0, stream>>>(x, xH, xL,
                                                                 (int)XN);
  hipMemsetAsync(h32, 0, 8 * BH * sizeof(float), stream);
  hipMemsetAsync(hH, 0, 8 * BH * sizeof(bf16_t), stream);
  hipMemsetAsync(hL, 0, 8 * BH * sizeof(bf16_t), stream);

  const dim3 sgrid(32, 2, 1);
  const dim3 ggrid(SEQ * BATCH / 16, 24, 1);
  float* dout = (float*)d_out;

  if (big) {
    // ---- phase 1: all input-side gates for layer 0 (parallel GEMMs) ----
    gi_gemm_kernel<<<ggrid, 128, 0, stream>>>(xH, xL, INDIM, 0, whi[0], wlo[0],
                                              bih_f0, Gf);
    gi_gemm_kernel<<<ggrid, 128, 0, stream>>>(xH, xL, INDIM, 1, whi[2], wlo[2],
                                              bih_b0, Gb);
    // ---- layer 0 scan (gh only) ----
    for (int t = 0; t < SEQ; ++t) {
      bf16_t* oH = o0H + (size_t)t * BATCH * 2 * HID;
      bf16_t* oL = o0L + (size_t)t * BATCH * 2 * HID;
      gru_step_kernel<true, INDIM><<<sgrid, 128, 0, stream>>>(
          nullptr, nullptr, Gf + (size_t)t * BATCH * G3,
          Gb + (size_t)t * BATCH * G3, whi[0], wlo[0], whi[1], wlo[1], bih_f0,
          bhh_f0, whi[2], wlo[2], whi[3], wlo[3], bih_b0, bhh_b0, H32(0, t & 1),
          HHp(0, t & 1), HLp(0, t & 1), H32(0, (t + 1) & 1),
          HHp(0, (t + 1) & 1), HLp(0, (t + 1) & 1), H32(1, t & 1),
          HHp(1, t & 1), HLp(1, t & 1), H32(1, (t + 1) & 1),
          HHp(1, (t + 1) & 1), HLp(1, (t + 1) & 1), nullptr, oH, oL, 2 * HID);
    }
    // ---- phase 2: layer-1 input-side gates (reuse G buffers) ----
    gi_gemm_kernel<<<ggrid, 128, 0, stream>>>(o0H, o0L, 2 * HID, 0, whi[4],
                                              wlo[4], bih_f1, Gf);
    gi_gemm_kernel<<<ggrid, 128, 0, stream>>>(o0H, o0L, 2 * HID, 1, whi[6],
                                              wlo[6], bih_b1, Gb);
    // ---- layer 1 scan ----
    for (int t = 0; t < SEQ; ++t) {
      float* ot = dout + (size_t)t * BATCH * 2 * HID;
      gru_step_kernel<true, INDIM><<<sgrid, 128, 0, stream>>>(
          nullptr, nullptr, Gf + (size_t)t * BATCH * G3,
          Gb + (size_t)t * BATCH * G3, whi[4], wlo[4], whi[5], wlo[5], bih_f1,
          bhh_f1, whi[6], wlo[6], whi[7], wlo[7], bih_b1, bhh_b1, H32(2, t & 1),
          HHp(2, t & 1), HLp(2, t & 1), H32(2, (t + 1) & 1),
          HHp(2, (t + 1) & 1), HLp(2, (t + 1) & 1), H32(3, t & 1),
          HHp(3, t & 1), HLp(3, t & 1), H32(3, (t + 1) & 1),
          HHp(3, (t + 1) & 1), HLp(3, (t + 1) & 1), ot, nullptr, nullptr,
          2 * HID);
    }
  } else {
    // ---- fallback: fused gi+gh per step (smaller workspace) ----
    for (int t = 0; t < SEQ; ++t) {
      bf16_t* oH = o0H + (size_t)t * BATCH * 2 * HID;
      bf16_t* oL = o0L + (size_t)t * BATCH * 2 * HID;
      gru_step_kernel<false, INDIM><<<sgrid, 128, 0, stream>>>(
          xH + (size_t)t * BATCH * INDIM, xL + (size_t)t * BATCH * INDIM,
          nullptr, nullptr, whi[0], wlo[0], whi[1], wlo[1], bih_f0, bhh_f0,
          whi[2], wlo[2], whi[3], wlo[3], bih_b0, bhh_b0, H32(0, t & 1),
          HHp(0, t & 1), HLp(0, t & 1), H32(0, (t + 1) & 1),
          HHp(0, (t + 1) & 1), HLp(0, (t + 1) & 1), H32(1, t & 1),
          HHp(1, t & 1), HLp(1, t & 1), H32(1, (t + 1) & 1),
          HHp(1, (t + 1) & 1), HLp(1, (t + 1) & 1), nullptr, oH, oL, 2 * HID);
    }
    for (int t = 0; t < SEQ; ++t) {
      float* ot = dout + (size_t)t * BATCH * 2 * HID;
      gru_step_kernel<false, 2 * HID><<<sgrid, 128, 0, stream>>>(
          o0H + (size_t)t * BATCH * 2 * HID, o0L + (size_t)t * BATCH * 2 * HID,
          nullptr, nullptr, whi[4], wlo[4], whi[5], wlo[5], bih_f1, bhh_f1,
          whi[6], wlo[6], whi[7], wlo[7], bih_b1, bhh_b1, H32(2, t & 1),
          HHp(2, t & 1), HLp(2, t & 1), H32(2, (t + 1) & 1),
          HHp(2, (t + 1) & 1), HLp(2, (t + 1) & 1), H32(3, t & 1),
          HHp(3, t & 1), HLp(3, t & 1), H32(3, (t + 1) & 1),
          HHp(3, (t + 1) & 1), HLp(3, (t + 1) & 1), ot, nullptr, nullptr,
          2 * HID);
    }
  }
}
